// Mamba2Block_10015863734462
// MI455X (gfx1250) — matrix operand order repaired and checked by reading, not compiled, not run on hardware
//
#include <hip/hip_runtime.h>
#include <hip/hip_bf16.h>
#include <math.h>

// ---------------- CDNA5 WMMA helpers (gfx1250, wave32) ----------------
typedef __attribute__((ext_vector_type(16))) _Float16 v16h;
typedef __attribute__((ext_vector_type(8)))  _Float16 v8h;
typedef __attribute__((ext_vector_type(8)))  float    v8f;

__device__ __forceinline__ v8f wmma_f16(v16h a, v16h b, v8f c) {
  // 8 args: (neg_a, A, neg_b, B, c_mod, C, reuse_a, reuse_b)
  return __builtin_amdgcn_wmma_f32_16x16x32_f16(false, a, false, b, (short)0, c, false, false);
}

__device__ __forceinline__ v16h cat16(v8h lo, v8h hi) {
  return __builtin_shufflevector(lo, hi, 0, 1, 2, 3, 4, 5, 6, 7, 8, 9, 10, 11, 12, 13, 14, 15);
}

// A-fragment (16x32) from row-major [M][K] LDS tile, leading dim `ld`.
// lane m = lane%16, k-half = lane/16 -> two aligned 128-bit LDS loads.
__device__ __forceinline__ v16h ldA(const _Float16* s, int row0, int ld, int koff, int lane) {
  int m = lane & 15, kh = lane >> 4;
  const _Float16* p = s + (row0 + m) * ld + koff + kh * 8;
  return cat16(*(const v8h*)p, *(const v8h*)(p + 16));
}

// B-fragment (32x16) from K-innermost LDS tile: element [k][col] = s[(col0+col)*ldT + koff + k].
__device__ __forceinline__ v16h ldBT(const _Float16* s, int col0, int ldT, int koff, int lane) {
  int n = lane & 15, kh = lane >> 4;
  const _Float16* p = s + (col0 + n) * ldT + koff + kh * 8;
  return cat16(*(const v8h*)p, *(const v8h*)(p + 16));
}

__device__ __forceinline__ float silu(float x) { return x / (1.f + expf(-x)); }

// ---------------- model dims ----------------
constexpr int D_MODEL = 1024;
constexpr int D_INNER = 2048;
constexpr int NHEADS  = 32;
constexpr int HEADDIM = 64;
constexpr int D_STATE = 64;
constexpr int CHUNK   = 128;
constexpr int NCHUNK  = 16;
constexpr int BATCH   = 2;
constexpr int SEQLEN  = 2048;
constexpr int ROWS    = BATCH * SEQLEN;                 // 4096
constexpr int PROJ_W  = D_INNER + NHEADS + 2 * D_STATE; // 2208
constexpr int PROJ_LD = 2304;                           // padded to multiple of 128
constexpr int OFF_DT  = D_INNER;                        // 2048
constexpr int OFF_B   = D_INNER + NHEADS;               // 2080
constexpr int OFF_C   = D_INNER + NHEADS + D_STATE;     // 2144

// ---------------- pad W_xproj columns 2208 -> 2304 ----------------
__global__ void pad_w(const float* __restrict__ W, float* __restrict__ Wp) {
  int idx = blockIdx.x * 256 + threadIdx.x;   // D_INNER*PROJ_LD threads
  int n = idx % PROJ_LD;
  int k = idx / PROJ_LD;
  Wp[idx] = (n < PROJ_W) ? W[(size_t)k * PROJ_W + n] : 0.f;
}

// ---------------- generic f32-in f32-out WMMA GEMM (no guards) ----------------
// C[M,N] = A[M,K] * B[K,N]; M%64==0, N%128==0, K%32==0.
// Block: 256 threads = 8 waves in a 2x4 grid; each wave computes a 32x32 quad
// (2x2 WMMA tiles) -> block tile 64x128, 4 WMMAs per wave per k-step.
__global__ void gemm_wmma(const float* __restrict__ A, const float* __restrict__ B,
                          float* __restrict__ C, int M, int N, int K) {
  __shared__ __align__(16) _Float16 As[64 * 32];     // [m][k]
  __shared__ __align__(16) _Float16 BsT[128 * 32];   // [n][k]  (K innermost)
  int tid = threadIdx.x, lane = tid & 31, w = tid >> 5;
  int wr = w >> 2, wc = w & 3;            // wave row (2) / col (4)
  int m_base = blockIdx.y * 64;
  int n_base = blockIdx.x * 128;
  v8f acc00 = {}, acc01 = {}, acc10 = {}, acc11 = {};
  for (int kk = 0; kk < K; kk += 32) {
    // A tile 64x32: 8 contiguous floats/thread -> vector loads, packed LDS stores
#pragma unroll
    for (int i = 0; i < 8; ++i) {
      int e = tid * 8 + i;
      int r = e >> 5, kc = e & 31;
      As[e] = (_Float16)A[(size_t)(m_base + r) * K + kk + kc];
    }
    // B tile 32x128: 16 contiguous floats/thread (coalesced), transposed LDS scatter
#pragma unroll
    for (int i = 0; i < 16; ++i) {
      int e = tid * 16 + i;
      int kr = e >> 7, nc = e & 127;
      BsT[nc * 32 + kr] = (_Float16)B[(size_t)(kk + kr) * N + n_base + nc];
    }
    if (kk + 32 < K) {
      __builtin_prefetch(&A[(size_t)(m_base + (tid >> 2)) * K + kk + 32], 0, 1);
      __builtin_prefetch(&B[(size_t)(kk + 32 + (tid >> 5)) * N + n_base], 0, 1);
    }
    __syncthreads();
    v16h a0 = ldA(As, wr * 32, 32, 0, lane);
    v16h a1 = ldA(As, wr * 32 + 16, 32, 0, lane);
    v16h b0 = ldBT(BsT, wc * 32, 32, 0, lane);
    v16h b1 = ldBT(BsT, wc * 32 + 16, 32, 0, lane);
    acc00 = wmma_f16(a0, b0, acc00);
    acc01 = wmma_f16(a0, b1, acc01);
    acc10 = wmma_f16(a1, b0, acc10);
    acc11 = wmma_f16(a1, b1, acc11);
    __syncthreads();
  }
  int nc = lane & 15, roff = (lane >> 4) << 3;
  int m0 = m_base + wr * 32, n0 = n_base + wc * 32;
#pragma unroll
  for (int r = 0; r < 8; ++r) {
    C[(size_t)(m0 + roff + r) * N + n0 + nc]           = acc00[r];
    C[(size_t)(m0 + roff + r) * N + n0 + 16 + nc]      = acc01[r];
    C[(size_t)(m0 + 16 + roff + r) * N + n0 + nc]      = acc10[r];
    C[(size_t)(m0 + 16 + roff + r) * N + n0 + 16 + nc] = acc11[r];
  }
}

// ---------------- depthwise causal conv (W=4) + SiLU ----------------
__global__ void conv_silu(const float* __restrict__ xz, const float* __restrict__ cw,
                          const float* __restrict__ cb, float* __restrict__ xc) {
  int idx = blockIdx.x * 256 + threadIdx.x;   // ROWS*D_INNER threads
  int row = idx >> 11;
  int d   = idx & 2047;
  int b = row >> 11, t = row & 2047;
  float s = cb[d];
#pragma unroll
  for (int j = 0; j < 4; ++j) {
    int tt = t - 3 + j;
    if (tt >= 0) s += xz[(size_t)(b * SEQLEN + tt) * (2 * D_INNER) + d] * cw[d * 4 + j];
  }
  xc[(size_t)row * D_INNER + d] = silu(s);
}

// ---------------- dt = softplus(dt_raw @ W_dt + b_dt) ----------------
__global__ void dt_kernel(const float* __restrict__ proj, const float* __restrict__ Wdt,
                          const float* __restrict__ bdt, float* __restrict__ dt) {
  int idx = blockIdx.x * 256 + threadIdx.x;   // ROWS*NHEADS
  int row = idx >> 5, h = idx & 31;
  const float* pr = proj + (size_t)row * PROJ_LD + OFF_DT;
  float s = bdt[h];
#pragma unroll 8
  for (int j = 0; j < NHEADS; ++j) s += pr[j] * Wdt[j * NHEADS + h];
  dt[(size_t)row * NHEADS + h] = (s > 20.f) ? s : log1pf(expf(s));
}

// ---------------- SSD pass 1: cumsum + per-chunk local states ----------------
// grid = BATCH*NCHUNK*NHEADS = 1024 ; block = 256 (8 waves)
__global__ void ssd_chunk_states(const float* __restrict__ proj, const float* __restrict__ dt,
                                 const float* __restrict__ A_log,
                                 float* __restrict__ wsAcs, float* __restrict__ wsSloc) {
  __shared__ float acs[CHUNK];
  __shared__ __align__(16) _Float16 XdT[HEADDIM * CHUNK];  // [p][l]  (K=l innermost)
  __shared__ __align__(16) _Float16 BdT[D_STATE * CHUNK];  // [n][l]  (B * decay)
  int bid = blockIdx.x;
  int h = bid & 31, c = (bid >> 5) & 15, b = bid >> 9;
  int tid = threadIdx.x, lane = tid & 31, w = tid >> 5;
  int rowbase = b * SEQLEN + c * CHUNK;
  float Ah = -expf(A_log[h]);
  if (tid < CHUNK) acs[tid] = Ah * dt[(size_t)(rowbase + tid) * NHEADS + h];
  __syncthreads();
  for (int off = 1; off < CHUNK; off <<= 1) {   // inclusive scan
    float add = 0.f;
    if (tid < CHUNK && tid >= off) add = acs[tid - off];
    __syncthreads();
    if (tid < CHUNK) acs[tid] += add;
    __syncthreads();
  }
  float Asum = acs[CHUNK - 1];
  if (tid < CHUNK) wsAcs[(size_t)bid * CHUNK + tid] = acs[tid];
  // coalesced global reads, transposed LDS scatter
  for (int i = tid; i < CHUNK * HEADDIM; i += 256) {
    int l = i >> 6, p = i & 63;
    int row = rowbase + l;
    XdT[p * CHUNK + l] = (_Float16)(proj[(size_t)row * PROJ_LD + h * HEADDIM + p] *
                                    dt[(size_t)row * NHEADS + h]);
  }
  for (int i = tid; i < CHUNK * D_STATE; i += 256) {
    int l = i >> 6, n = i & 63;
    int row = rowbase + l;
    BdT[n * CHUNK + l] = (_Float16)(proj[(size_t)row * PROJ_LD + OFF_B + n] *
                                    expf(Asum - acs[l]));
  }
  __syncthreads();
  // S_loc[n][p] = sum_l BdT[n][l] * Xd[l][p] : 64x64, K=128 -> 16 tiles / 8 waves
  for (int tt = w; tt < 16; tt += 8) {
    int nt = tt >> 2, pt = tt & 3;
    v8f acc = {};
#pragma unroll
    for (int ks = 0; ks < 4; ++ks) {
      v16h a  = ldA(BdT, nt * 16, CHUNK, ks * 32, lane);
      v16h bb = ldBT(XdT, pt * 16, CHUNK, ks * 32, lane);
      acc = wmma_f16(a, bb, acc);
    }
    int nc = lane & 15, roff = (lane >> 4) << 3;
    float* out = wsSloc + (size_t)bid * (D_STATE * HEADDIM);
#pragma unroll
    for (int r = 0; r < 8; ++r)
      out[(nt * 16 + roff + r) * HEADDIM + pt * 16 + nc] = acc[r];
  }
}

// ---------------- SSD pass 2: inter-chunk recurrence ----------------
__global__ void ssd_scan(const float* __restrict__ wsSloc, const float* __restrict__ wsAcs,
                         float* __restrict__ wsSprior) {
  int bh = blockIdx.x;
  int b = bh >> 5, h = bh & 31;
  int tid = threadIdx.x;
  float S[16];
#pragma unroll
  for (int j = 0; j < 16; ++j) S[j] = 0.f;
  for (int c = 0; c < NCHUNK; ++c) {
    int sidx = (b * NCHUNK + c) * NHEADS + h;
    float dec = expf(wsAcs[(size_t)sidx * CHUNK + CHUNK - 1]);
    const float* sl = wsSloc + (size_t)sidx * 4096;
    float* sp = wsSprior + (size_t)sidx * 4096;
#pragma unroll
    for (int j = 0; j < 16; ++j) {
      sp[tid * 16 + j] = S[j];
      S[j] = dec * S[j] + sl[tid * 16 + j];
    }
  }
}

// ---------------- SSD pass 3: Y = Y_diag + Y_off ; fuse D-skip + SiLU gate ----
// grid = 1024 ; block = 256 (wave w owns l-tile w)
__global__ void ssd_output(const float* __restrict__ proj, const float* __restrict__ dt,
                           const float* __restrict__ xz, const float* __restrict__ wsAcs,
                           const float* __restrict__ wsSprior, const float* __restrict__ Dp,
                           float* __restrict__ Yg) {
  __shared__ __align__(16) _Float16 Ce[CHUNK * D_STATE];  // [l][n] * exp(+Acs[l])
  __shared__ __align__(16) _Float16 Be[CHUNK * D_STATE];  // [s][n] * exp(-Acs[s])
  __shared__ __align__(16) _Float16 U[HEADDIM * CHUNK];   // SpT[p][n] (64x64) then XdT[p][l]
  __shared__ __align__(16) _Float16 G[CHUNK * 32];        // masked-decay panel [l][s-local]
  __shared__ float acs[CHUNK];
  int bid = blockIdx.x;
  int h = bid & 31, c = (bid >> 5) & 15, b = bid >> 9;
  int tid = threadIdx.x, lane = tid & 31, w = tid >> 5;
  int rowbase = b * SEQLEN + c * CHUNK;
  if (tid < CHUNK) acs[tid] = wsAcs[(size_t)bid * CHUNK + tid];
  __syncthreads();
  for (int i = tid; i < CHUNK * D_STATE; i += 256) {
    int l = i >> 6, n = i & 63;
    int row = rowbase + l;
    Ce[i] = (_Float16)(proj[(size_t)row * PROJ_LD + OFF_C + n] * expf(acs[l]));
    Be[i] = (_Float16)(proj[(size_t)row * PROJ_LD + OFF_B + n] * expf(-acs[l]));
  }
  // S_prior[n][p] -> SpT[p][n] (K=n innermost)
  for (int i = tid; i < D_STATE * HEADDIM; i += 256) {
    int n = i >> 6, p = i & 63;
    U[p * 64 + n] = (_Float16)wsSprior[(size_t)bid * 4096 + i];
  }
  __syncthreads();
  // Y_off = Ce (128x64) @ S_prior (64x64)
  v8f Yacc[4];
#pragma unroll
  for (int pt = 0; pt < 4; ++pt) {
    v8f acc = {};
#pragma unroll
    for (int ks = 0; ks < 2; ++ks) {
      v16h a  = ldA(Ce, w * 16, D_STATE, ks * 32, lane);
      v16h bb = ldBT(U, pt * 16, 64, ks * 32, lane);
      acc = wmma_f16(a, bb, acc);
    }
    Yacc[pt] = acc;
  }
  __syncthreads();
  // overwrite U with XdT[p][l] = (X * dt)^T
  for (int i = tid; i < CHUNK * HEADDIM; i += 256) {
    int l = i >> 6, p = i & 63;
    int row = rowbase + l;
    U[p * CHUNK + l] = (_Float16)(proj[(size_t)row * PROJ_LD + h * HEADDIM + p] *
                                  dt[(size_t)row * NHEADS + h]);
  }
  __syncthreads();
  // Y_diag: 4 panels of s-width 32
  for (int sp = 0; sp < 4; ++sp) {
    int s0 = sp * 32;
#pragma unroll
    for (int st = 0; st < 2; ++st) {   // wave: G tiles (l-tile w, s-tiles 0/1)
      v8f g = {};
#pragma unroll
      for (int ks = 0; ks < 2; ++ks) {
        v16h a  = ldA(Ce, w * 16, D_STATE, ks * 32, lane);
        v16h bb = ldBT(Be, s0 + st * 16, D_STATE, ks * 32, lane);  // Be^T: [k=n][col=s]
        g = wmma_f16(a, bb, g);
      }
      int nc = lane & 15, roff = (lane >> 4) << 3;
#pragma unroll
      for (int r = 0; r < 8; ++r) {
        int l = w * 16 + roff + r;
        int s = s0 + st * 16 + nc;
        G[l * 32 + st * 16 + nc] = (_Float16)((s <= l) ? g[r] : 0.f);
      }
    }
    __syncthreads();
#pragma unroll
    for (int pt = 0; pt < 4; ++pt) {   // Y += G_panel (128x32) @ Xd[s0:s0+32][:]
      v16h a  = ldA(G, w * 16, 32, 0, lane);
      v16h bb = ldBT(U, pt * 16, CHUNK, s0, lane);
      Yacc[pt] = wmma_f16(a, bb, Yacc[pt]);
    }
    __syncthreads();
  }
  // epilogue: + X*D, gate with silu(z)
  float Dh = Dp[h];
  int nc = lane & 15, roff = (lane >> 4) << 3;
#pragma unroll
  for (int pt = 0; pt < 4; ++pt) {
#pragma unroll
    for (int r = 0; r < 8; ++r) {
      int l = w * 16 + roff + r;
      int p = pt * 16 + nc;
      int row = rowbase + l;
      float x = proj[(size_t)row * PROJ_LD + h * HEADDIM + p];
      float y = Yacc[pt][r] + x * Dh;
      float z = xz[(size_t)row * (2 * D_INNER) + D_INNER + h * HEADDIM + p];
      Yg[(size_t)row * D_INNER + h * HEADDIM + p] = y * silu(z);
    }
  }
}

// ---------------- launcher ----------------
extern "C" void kernel_launch(void* const* d_in, const int* in_sizes, int n_in,
                              void* d_out, int out_size, void* d_ws, size_t ws_size,
                              hipStream_t stream) {
  (void)in_sizes; (void)n_in; (void)out_size; (void)ws_size;
  const float* x      = (const float*)d_in[0];
  const float* W_in   = (const float*)d_in[1];
  const float* conv_w = (const float*)d_in[2];
  const float* conv_b = (const float*)d_in[3];
  const float* W_xprj = (const float*)d_in[4];
  const float* W_dt   = (const float*)d_in[5];
  const float* b_dt   = (const float*)d_in[6];
  const float* A_log  = (const float*)d_in[7];
  const float* Dvec   = (const float*)d_in[8];
  const float* W_out  = (const float*)d_in[9];
  float* out = (float*)d_out;

  float* ws      = (float*)d_ws;
  float* xz      = ws;                                   // 4096*4096
  float* xc      = xz   + (size_t)ROWS * 2 * D_INNER;    // 4096*2048
  float* proj    = xc   + (size_t)ROWS * D_INNER;        // 4096*2304
  float* dtb     = proj + (size_t)ROWS * PROJ_LD;        // 4096*32
  float* acs     = dtb  + (size_t)ROWS * NHEADS;         // 1024*128
  float* sloc    = acs  + (size_t)BATCH * NCHUNK * NHEADS * CHUNK;  // 1024*4096
  float* sprior  = sloc + (size_t)BATCH * NCHUNK * NHEADS * 4096;   // 1024*4096
  float* yg      = sprior + (size_t)BATCH * NCHUNK * NHEADS * 4096; // 4096*2048
  float* wpad    = yg + (size_t)ROWS * D_INNER;          // 2048*2304

  dim3 blk(256);
  // 0) pad W_xproj to 2304 columns
  pad_w<<<(D_INNER * PROJ_LD) / 256, blk, 0, stream>>>(W_xprj, wpad);
  // 1) xz = x @ W_in
  gemm_wmma<<<dim3((2 * D_INNER) / 128, ROWS / 64), blk, 0, stream>>>(
      x, W_in, xz, ROWS, 2 * D_INNER, D_MODEL);
  // 2) depthwise conv + silu
  conv_silu<<<(ROWS * D_INNER) / 256, blk, 0, stream>>>(xz, conv_w, conv_b, xc);
  // 3) proj = xc @ W_xproj (padded, guard-free)
  gemm_wmma<<<dim3(PROJ_LD / 128, ROWS / 64), blk, 0, stream>>>(
      xc, wpad, proj, ROWS, PROJ_LD, D_INNER);
  // 4) dt
  dt_kernel<<<(ROWS * NHEADS) / 256, blk, 0, stream>>>(proj, W_dt, b_dt, dtb);
  // 5) per-chunk cumsums + local states
  ssd_chunk_states<<<BATCH * NCHUNK * NHEADS, blk, 0, stream>>>(proj, dtb, A_log, acs, sloc);
  // 6) inter-chunk scan
  ssd_scan<<<BATCH * NHEADS, blk, 0, stream>>>(sloc, acs, sprior);
  // 7) Y_diag + Y_off + skip + gate
  ssd_output<<<BATCH * NCHUNK * NHEADS, blk, 0, stream>>>(proj, dtb, xz, acs, sprior, Dvec, yg);
  // 8) out = Yg @ W_out
  gemm_wmma<<<dim3(D_MODEL / 128, ROWS / 64), blk, 0, stream>>>(
      yg, W_out, out, ROWS, D_MODEL, D_INNER);
}
